// MLA_83270825935232
// MI455X (gfx1250) — matrix operand order repaired and checked by reading, not compiled, not run on hardware
//
#include <hip/hip_runtime.h>

// ---------------- MLA constants (from reference) ----------------
constexpr int Bb   = 2;
constexpr int Ss   = 2048;
constexpr int DIMc = 2048;
constexpr int Hh   = 16;
constexpr int QLR  = 1536;
constexpr int KVLR = 512;
constexpr int DN   = 128;   // D_NOPE
constexpr int DR   = 64;    // D_ROPE
constexpr int DQK  = 192;
constexpr int DV   = 128;
constexpr int MT   = Bb * Ss;           // 4096 tokens total
constexpr int DEXT = KVLR + DR;         // 576 extended latent width
constexpr float SCALE = 0.07216878364870322f;  // 192^-0.5

#ifndef __has_builtin
#define __has_builtin(x) 0
#endif
#if __has_builtin(__builtin_amdgcn_global_load_async_to_lds_b128) && \
    __has_builtin(__builtin_amdgcn_s_wait_asynccnt)
#define USE_ASYNC_COPY 1
#else
#define USE_ASYNC_COPY 0
#endif

// ---------------- types / helpers ----------------
typedef __attribute__((ext_vector_type(16))) __bf16        bf16x16;
typedef __attribute__((ext_vector_type(8)))  float         f32x8;
typedef __attribute__((ext_vector_type(8)))  unsigned int  u32x8;
typedef __attribute__((ext_vector_type(4)))  unsigned int  u32x4;
typedef __attribute__((vector_size(16)))     int           v4i_t;
#if defined(__AMDGCN__)
typedef __attribute__((address_space(1))) v4i_t* gv4i_p;   // global (device) AS
typedef __attribute__((address_space(3))) v4i_t* lv4i_p;   // LDS AS
#endif

union U8 { u32x8 v; u32x4 q[2]; };

#define F8Z {0.f,0.f,0.f,0.f,0.f,0.f,0.f,0.f}

__device__ __forceinline__ unsigned short f2bf(float f) {
  unsigned int u = __builtin_bit_cast(unsigned int, f);
  u += 0x7FFFu + ((u >> 16) & 1u);           // round-to-nearest-even
  return (unsigned short)(u >> 16);
}

__device__ __forceinline__ f32x8 wmma_bf16(u32x8 a, u32x8 b, f32x8 c) {
  return __builtin_amdgcn_wmma_f32_16x16x32_bf16(
      false, __builtin_bit_cast(bf16x16, a),
      false, __builtin_bit_cast(bf16x16, b),
      (short)0, c, false, false);
}

// ---------------- f32 -> bf16 cast (vectorized) ----------------
__global__ void cast_f32_bf16(const float* __restrict__ in,
                              unsigned short* __restrict__ out, long n4) {
  long i = (long)blockIdx.x * blockDim.x + threadIdx.x;
  long stride = (long)gridDim.x * blockDim.x;
  const float4* in4 = (const float4*)in;
  uint2* out2 = (uint2*)out;
  for (; i < n4; i += stride) {
    float4 v = in4[i];
    uint2 r;
    r.x = (unsigned)f2bf(v.x) | ((unsigned)f2bf(v.y) << 16);
    r.y = (unsigned)f2bf(v.z) | ((unsigned)f2bf(v.w) << 16);
    out2[i] = r;
  }
}

// ---------------- LDS-tiled double-buffered bf16 WMMA GEMM ----------------
// C[m,n] = sum_k A[m,k]*B[k,n] per head z:
//   A elem: A[aHead*z + m*lda + k]
//   B elem: Bm[bBase + bHead*z + k*bsk + n*bsn]   (requires bsn==1 or bsk==1)
//   C elem: Cp[cHead*z + m*ldc + n]   (f32 or bf16)
// BM=128, BN=64, BK=32; 256 threads = 8 waves; wave w -> rows 16w..16w+15.
// A tile is copied global->LDS with gfx1250 async-to-LDS ops (ASYNCcnt).
__global__ __launch_bounds__(256) void gemm_bf16(
    const unsigned short* __restrict__ A, long lda, long aHead,
    const unsigned short* __restrict__ Bm, long bsk, long bsn, long bBase, long bHead,
    void* __restrict__ Cp, long ldc, long cHead, int cBf16, int K)
{
  const int tid = threadIdx.x, lane = tid & 31, wave = tid >> 5;
  const int m0 = blockIdx.y * 128;
  const int n0 = blockIdx.x * 64;
  const long hA = (long)blockIdx.z * aHead;
  const long hB = bBase + (long)blockIdx.z * bHead;
  const long hC = (long)blockIdx.z * cHead;

  __shared__ __align__(16) unsigned short sA[2][128 * 32];   // [m][k] stride 32
  __shared__ __align__(16) unsigned short sB[2][64 * 34];    // [n][k] stride 34 (pad)

  const int aM  = lane & 15;             // fragment row
  const int kh8 = (lane >> 4) * 8;       // A K sub-offset / C row sub-offset
  const int bN  = lane & 15;             // fragment col
  const int bKb = (lane >> 4) * 16;      // B K sub-offset
  const int mBase = m0 + wave * 16;

  // ---- A tile movers ----
#if USE_ASYNC_COPY
  auto loadA = [&](int k0, int buf) {    // async global->LDS, no VGPR staging
#pragma unroll
    for (int c = 0; c < 2; ++c) {
      int ch = tid + c * 256;            // 512 chunks: 128 rows x 4 x 16B
      int row = ch >> 2, col8 = (ch & 3) * 8;
      const unsigned short* g = A + hA + (long)(m0 + row) * lda + k0 + col8;
      __builtin_amdgcn_global_load_async_to_lds_b128(
          (gv4i_p)g, (lv4i_p)&sA[buf][row * 32 + col8], 0, 0);
    }
  };
#else
  uint4 arg[2];
  auto gloadA = [&](int k0) {
#pragma unroll
    for (int c = 0; c < 2; ++c) {
      int ch = tid + c * 256;
      int row = ch >> 2, col8 = (ch & 3) * 8;
      arg[c] = *(const uint4*)(A + hA + (long)(m0 + row) * lda + k0 + col8);
    }
  };
  auto stolA = [&](int buf) {
#pragma unroll
    for (int c = 0; c < 2; ++c) {
      int ch = tid + c * 256;
      int row = ch >> 2, col8 = (ch & 3) * 8;
      *(uint4*)&sA[buf][row * 32 + col8] = arg[c];
    }
  };
#endif

  // ---- B tile movers (VGPR-staged; bsn==1 path transposes on LDS store) ----
  uint4 brg;
  auto gloadB = [&](int k0) {
    if (bsn == 1) {                      // 256 chunks: 32 k-rows x 8 x 16B
      int k = tid >> 3, n8 = (tid & 7) * 8;
      brg = *(const uint4*)(Bm + hB + (long)(k0 + k) * bsk + n0 + n8);
    } else {                             // bsk==1: 64 n-rows x 4 x 16B
      int n = tid >> 2, k8 = (tid & 3) * 8;
      brg = *(const uint4*)(Bm + hB + k0 + k8 + (long)(n0 + n) * bsn);
    }
  };
  auto stolB = [&](int buf) {
    if (bsn == 1) {                      // transpose on LDS store
      int k = tid >> 3, n8 = (tid & 7) * 8;
      union { uint4 v; unsigned short s[8]; } u; u.v = brg;
#pragma unroll
      for (int j = 0; j < 8; ++j) sB[buf][(n8 + j) * 34 + k] = u.s[j];
    } else {                             // already [n][k]-major
      int n = tid >> 2, k8 = (tid & 3) * 8;
      union { uint4 v; unsigned int w[4]; } u; u.v = brg;
#pragma unroll
      for (int j = 0; j < 4; ++j)
        *(unsigned int*)&sB[buf][n * 34 + k8 + 2 * j] = u.w[j];
    }
  };

  f32x8 acc[4] = {F8Z, F8Z, F8Z, F8Z};
  const int NT = K >> 5;
#if USE_ASYNC_COPY
  loadA(0, 0);
#else
  gloadA(0);
#endif
  gloadB(0);
  int buf = 0;
  for (int kt = 0; kt < NT; ++kt) {
#if !USE_ASYNC_COPY
    stolA(buf);
#endif
    stolB(buf);
#if USE_ASYNC_COPY
    __builtin_amdgcn_s_wait_asynccnt(0);   // this wave's A-tile copies landed
#endif
    __syncthreads();                       // all waves' tiles visible
    if (kt + 1 < NT) {
#if USE_ASYNC_COPY
      loadA((kt + 1) * 32, buf ^ 1);
#else
      gloadA((kt + 1) * 32);
#endif
      gloadB((kt + 1) * 32);
    }
    // fragments: load everything first, then 4 back-to-back WMMAs
    U8 au;
    au.q[0] = *(const u32x4*)&sA[buf][(wave * 16 + aM) * 32 + 2 * kh8];
    au.q[1] = *(const u32x4*)&sA[buf][(wave * 16 + aM) * 32 + 2 * kh8 + 8];
    u32x8 bu[4];
#pragma unroll
    for (int nt = 0; nt < 4; ++nt)
#pragma unroll
      for (int v = 0; v < 8; ++v)
        bu[nt][v] = *(const unsigned int*)&sB[buf][(nt * 16 + bN) * 34 + bKb + 2 * v];
#pragma unroll
    for (int nt = 0; nt < 4; ++nt)
      acc[nt] = wmma_bf16(au.v, bu[nt], acc[nt]);
    buf ^= 1;
    __syncthreads();
  }
#pragma unroll
  for (int nt = 0; nt < 4; ++nt) {
#pragma unroll
    for (int r = 0; r < 8; ++r) {
      int m = r + kh8;                   // C layout: M = r + 8*(lane>>4)
      long off = hC + (long)(mBase + m) * ldc + n0 + nt * 16 + bN;
      if (cBf16) ((unsigned short*)Cp)[off] = f2bf(acc[nt][r]);
      else       ((float*)Cp)[off] = acc[nt][r];
    }
  }
}

// ---------------- RMS norm rows (f32 in -> bf16 out) ----------------
__global__ __launch_bounds__(256) void rmsnorm_rows(
    const float* __restrict__ in, const float* __restrict__ w,
    unsigned short* __restrict__ out, int C)
{
  const int row = blockIdx.x, tid = threadIdx.x;
  __shared__ float red[256];
  const float* r = in + (long)row * C;
  float ss = 0.f;
  for (int i = tid; i < C; i += 256) { float v = r[i]; ss += v * v; }
  red[tid] = ss; __syncthreads();
  for (int s = 128; s > 0; s >>= 1) {
    if (tid < s) red[tid] += red[tid + s];
    __syncthreads();
  }
  float rms = rsqrtf(red[0] / (float)C + 1e-6f);
  unsigned short* o = out + (long)row * C;
  for (int i = tid; i < C; i += 256) o[i] = f2bf(r[i] * rms * w[i]);
}

// ---------------- q post: split nope / rope(pe) ----------------
__global__ __launch_bounds__(64) void postq_kernel(
    const float* __restrict__ qf, const float* __restrict__ freqs,
    unsigned short* __restrict__ qnope, unsigned short* __restrict__ qext)
{
  const int mh = blockIdx.x;             // m*H + h
  const int rowm = mh >> 4;
  const int s = rowm & (Ss - 1);
  const int t = threadIdx.x;
  const float* q = qf + (long)mh * DQK;
  qnope[(long)mh * DN + 2 * t]     = f2bf(q[2 * t]);
  qnope[(long)mh * DN + 2 * t + 1] = f2bf(q[2 * t + 1]);
  if (t < 32) {
    float x0 = q[DN + 2 * t], x1 = q[DN + 2 * t + 1];
    float th = freqs[(long)s * 32 + t];
    float c = cosf(th), sn = sinf(th);
    unsigned short* o = qext + (long)mh * DEXT + KVLR;
    o[2 * t]     = f2bf(x0 * c - x1 * sn);
    o[2 * t + 1] = f2bf(x0 * sn + x1 * c);
  }
}

// ---------------- kv post: rmsnorm(c_kv) + rope(k_pe) -> kext ----------------
__global__ __launch_bounds__(256) void postkv_kernel(
    const float* __restrict__ kvf, const float* __restrict__ kvw,
    const float* __restrict__ freqs, unsigned short* __restrict__ kext)
{
  const int row = blockIdx.x, tid = threadIdx.x;
  const int s = row & (Ss - 1);
  __shared__ float red[256];
  const float* r = kvf + (long)row * DEXT;
  float ss = 0.f;
  for (int i = tid; i < KVLR; i += 256) { float v = r[i]; ss += v * v; }
  red[tid] = ss; __syncthreads();
  for (int st = 128; st > 0; st >>= 1) {
    if (tid < st) red[tid] += red[tid + st];
    __syncthreads();
  }
  float rms = rsqrtf(red[0] / (float)KVLR + 1e-6f);
  unsigned short* o = kext + (long)row * DEXT;
  for (int i = tid; i < KVLR; i += 256) o[i] = f2bf(r[i] * rms * kvw[i]);
  if (tid < 32) {
    float x0 = r[KVLR + 2 * tid], x1 = r[KVLR + 2 * tid + 1];
    float th = freqs[(long)s * 32 + tid];
    float c = cosf(th), sn = sinf(th);
    o[KVLR + 2 * tid]     = f2bf(x0 * c - x1 * sn);
    o[KVLR + 2 * tid + 1] = f2bf(x0 * sn + x1 * c);
  }
}

// ---------------- kext transpose: kT[b][c][t] = kext[b*Ss+t][c] ----------------
__global__ __launch_bounds__(256) void transpose_kext(
    const unsigned short* __restrict__ kext, unsigned short* __restrict__ kT)
{
  __shared__ unsigned short tile[32][33];
  const int b = blockIdx.z;
  const int c0 = blockIdx.y * 32, t0 = blockIdx.x * 32;
  const int tx = threadIdx.x & 31, ty = threadIdx.x >> 5;    // 32x8
  for (int i = ty; i < 32; i += 8)
    tile[i][tx] = kext[(long)(b * Ss + t0 + i) * DEXT + c0 + tx];
  __syncthreads();
  for (int i = ty; i < 32; i += 8)
    kT[((long)b * DEXT + c0 + i) * Ss + t0 + tx] = tile[tx][i];
}

// ---------------- flash attention over 576-wide latent ----------------
// qext: (MT,H,576) bf16; kext: (MT,576) bf16; kT: (B,576,Ss) bf16 (V^T = rows 0..511)
// ob: (MT,H,512) bf16. grid (Ss/16, H, B); block 256 = 8 waves.
__global__ __launch_bounds__(256) void mla_flash(
    const unsigned short* __restrict__ qext,
    const unsigned short* __restrict__ kext,
    const unsigned short* __restrict__ kT,
    unsigned short* __restrict__ ob)
{
  const int qtile = blockIdx.x, h = blockIdx.y, b = blockIdx.z;
  const int tid = threadIdx.x, lane = tid & 31, wave = tid >> 5;
  const int q0 = qtile * 16;
  const int bS = b * Ss;

  __shared__ float sS[16][132];
  __shared__ __align__(16) unsigned short sP[16][128];
  __shared__ __align__(16) unsigned short sQ[16 * DEXT];
  __shared__ float s_mnew[16], s_alpha[16], s_mrun[16], s_lrun[16];

  // preload Q tile (16 x 576) as b128 copies
  for (int i = tid; i < 16 * (DEXT / 8); i += 256) {
    int r = i / (DEXT / 8), c = (i % (DEXT / 8)) * 8;
    *(uint4*)&sQ[r * DEXT + c] =
        *(const uint4*)(qext + ((long)(bS + q0 + r) * Hh + h) * DEXT + c);
  }
  if (tid < 16) { s_mrun[tid] = -1e30f; s_lrun[tid] = 0.f; }

  const int aM  = lane & 15;
  const int kh8 = (lane >> 4) * 8;
  const int bN  = lane & 15;
  const int bKb = (lane >> 4) * 16;
  const unsigned short* vtb = kT + (long)b * DEXT * Ss;   // [c][t]

  f32x8 acc[4] = {F8Z, F8Z, F8Z, F8Z};
  __syncthreads();

  const int q_last = q0 + 15;
  for (int t0 = 0; t0 <= q_last; t0 += 128) {
    const int kt0 = t0 + 16 * wave;          // wave-uniform
    f32x8 sc = F8Z;
    if (kt0 <= q_last) {
      int keyRow = kt0 + bN; if (keyRow > Ss - 1) keyRow = Ss - 1;
      const unsigned short* kro = kext + (long)(bS + keyRow) * DEXT;
#pragma unroll 2
      for (int kf = 0; kf < DEXT; kf += 32) {
        U8 au, bu;
        au.q[0] = *(const u32x4*)&sQ[aM * DEXT + kf + 2 * kh8];
        au.q[1] = *(const u32x4*)&sQ[aM * DEXT + kf + 2 * kh8 + 8];
        bu.q[0] = *(const u32x4*)&kro[kf + bKb];
        bu.q[1] = *(const u32x4*)&kro[kf + bKb + 8];
        sc = wmma_bf16(au.v, bu.v, sc);
      }
    }
    // scale + causal mask + write scores to LDS
#pragma unroll
    for (int r = 0; r < 8; ++r) {
      int qm = r + kh8;
      int ki = kt0 + bN;
      float v = (kt0 <= q_last && ki <= q0 + qm) ? sc[r] * SCALE : -1e30f;
      sS[qm][16 * wave + bN] = v;
    }
    __syncthreads();
    // online-softmax stats (wave 0, one lane per query row)
    if (wave == 0 && lane < 16) {
      const int row = lane;
      float mx = s_mrun[row];
      for (int j = 0; j < 128; ++j) mx = fmaxf(mx, sS[row][j]);
      float al = __expf(s_mrun[row] - mx);
      float sum = 0.f;
      for (int j = 0; j < 128; ++j) sum += __expf(sS[row][j] - mx);
      s_lrun[row] = s_lrun[row] * al + sum;
      s_alpha[row] = al; s_mnew[row] = mx; s_mrun[row] = mx;
    }
    __syncthreads();
    // build P (bf16) slice + rescale accumulators
#pragma unroll
    for (int j = 0; j < 8; ++j) {
      int e = lane * 8 + j;
      int row = e >> 4, col = (e & 15) + 16 * wave;
      sP[row][col] = f2bf(__expf(sS[row][col] - s_mnew[row]));
    }
#pragma unroll
    for (int r = 0; r < 8; ++r) {
      float a = s_alpha[r + kh8];
      acc[0][r] *= a; acc[1][r] *= a; acc[2][r] *= a; acc[3][r] *= a;
    }
    __syncthreads();
    // O += P (16x128) @ V (128 x 64-per-wave); preload all fragments, then WMMAs
    int nkeys = q_last - t0 + 1; if (nkeys > 128) nkeys = 128;
    const int ktcnt = (nkeys + 31) / 32;
    for (int kt = 0; kt < ktcnt; ++kt) {
      U8 au;
      au.q[0] = *(const u32x4*)&sP[aM][kt * 32 + 2 * kh8];
      au.q[1] = *(const u32x4*)&sP[aM][kt * 32 + 2 * kh8 + 8];
      int kb0 = t0 + kt * 32 + bKb;
      if (kb0 > Ss - 16) kb0 = Ss - 16;      // masked tail stays in-bounds (P=0)
      U8 bu[4];
#pragma unroll
      for (int nt = 0; nt < 4; ++nt) {
        const int col = wave * 64 + nt * 16 + bN;
        bu[nt].q[0] = *(const u32x4*)&vtb[(long)col * Ss + kb0];
        bu[nt].q[1] = *(const u32x4*)&vtb[(long)col * Ss + kb0 + 8];
      }
#pragma unroll
      for (int nt = 0; nt < 4; ++nt)
        acc[nt] = wmma_bf16(au.v, bu[nt].v, acc[nt]);
    }
    __syncthreads();
  }
  // epilogue: divide by l, store bf16
#pragma unroll
  for (int r = 0; r < 8; ++r) {
    int qm = r + kh8;
    float inv = 1.0f / s_lrun[qm];
    long base = ((long)(bS + q0 + qm) * Hh + h) * KVLR + wave * 64 + bN;
    ob[base + 0]  = f2bf(acc[0][r] * inv);
    ob[base + 16] = f2bf(acc[1][r] * inv);
    ob[base + 32] = f2bf(acc[2][r] * inv);
    ob[base + 48] = f2bf(acc[3][r] * inv);
  }
}

// ---------------- host orchestration ----------------
extern "C" void kernel_launch(void* const* d_in, const int* in_sizes, int n_in,
                              void* d_out, int out_size, void* d_ws, size_t ws_size,
                              hipStream_t stream)
{
  (void)in_sizes; (void)n_in; (void)out_size; (void)ws_size;
  const float* x        = (const float*)d_in[0];
  const float* freqs    = (const float*)d_in[1];
  // d_in[2] = start_pos (0, unused)
  const float* wq_a     = (const float*)d_in[3];
  const float* q_norm_w = (const float*)d_in[4];
  const float* wq_b     = (const float*)d_in[5];
  const float* wkv_a    = (const float*)d_in[6];
  const float* kv_norm_w= (const float*)d_in[7];
  const float* wkv_b    = (const float*)d_in[8];
  const float* wo       = (const float*)d_in[9];
  float* out = (float*)d_out;

  char* p = (char*)d_ws;
  auto alloc = [&](size_t bytes) {
    char* r = p; p += (bytes + 255) & ~(size_t)255; return r;
  };
  unsigned short* xb     = (unsigned short*)alloc((size_t)MT * DIMc * 2);
  unsigned short* wqab   = (unsigned short*)alloc((size_t)DIMc * QLR * 2);
  unsigned short* wqbb   = (unsigned short*)alloc((size_t)QLR * (Hh * DQK) * 2);
  unsigned short* wkvab  = (unsigned short*)alloc((size_t)DIMc * DEXT * 2);
  unsigned short* wkvbb  = (unsigned short*)alloc((size_t)KVLR * (Hh * (DN + DV)) * 2);
  unsigned short* wob    = (unsigned short*)alloc((size_t)(Hh * DV) * DIMc * 2);
  float*          scr    = (float*)alloc((size_t)MT * (Hh * DQK) * 4);
  unsigned short* qlatb  = (unsigned short*)alloc((size_t)MT * QLR * 2);
  unsigned short* qnopeb = (unsigned short*)alloc((size_t)MT * Hh * DN * 2);
  unsigned short* qextb  = (unsigned short*)alloc((size_t)MT * Hh * DEXT * 2);
  unsigned short* kextb  = (unsigned short*)alloc((size_t)MT * DEXT * 2);
  unsigned short* kTb    = (unsigned short*)alloc((size_t)Bb * DEXT * Ss * 2);
  unsigned short* obuf   = (unsigned short*)alloc((size_t)MT * Hh * KVLR * 2);
  unsigned short* ocat   = (unsigned short*)alloc((size_t)MT * Hh * DV * 2);

  // 1) casts to bf16 (element counts all divisible by 4)
  cast_f32_bf16<<<1024, 256, 0, stream>>>(x,     xb,    (long)MT * DIMc / 4);
  cast_f32_bf16<<<1024, 256, 0, stream>>>(wq_a,  wqab,  (long)DIMc * QLR / 4);
  cast_f32_bf16<<<1024, 256, 0, stream>>>(wq_b,  wqbb,  (long)QLR * Hh * DQK / 4);
  cast_f32_bf16<<<1024, 256, 0, stream>>>(wkv_a, wkvab, (long)DIMc * DEXT / 4);
  cast_f32_bf16<<<1024, 256, 0, stream>>>(wkv_b, wkvbb, (long)KVLR * Hh * (DN + DV) / 4);
  cast_f32_bf16<<<1024, 256, 0, stream>>>(wo,    wob,   (long)(Hh * DV) * DIMc / 4);

  const dim3 blk(256);
  // 2) q_lat = x @ wq_a -> f32 scratch
  gemm_bf16<<<dim3(QLR / 64, MT / 128, 1), blk, 0, stream>>>(
      xb, DIMc, 0, wqab, QLR, 1, 0, 0, scr, QLR, 0, 0, DIMc);
  // 3) rmsnorm -> bf16
  rmsnorm_rows<<<MT, blk, 0, stream>>>(scr, q_norm_w, qlatb, QLR);
  // 4) q = q_lat_n @ wq_b -> f32 scratch
  gemm_bf16<<<dim3(Hh * DQK / 64, MT / 128, 1), blk, 0, stream>>>(
      qlatb, QLR, 0, wqbb, Hh * DQK, 1, 0, 0, scr, Hh * DQK, 0, 0, QLR);
  // 5) split q_nope / rope q_pe
  postq_kernel<<<MT * Hh, 64, 0, stream>>>(scr, freqs, qnopeb, qextb);
  // 6) kv = x @ wkv_a -> f32 scratch
  gemm_bf16<<<dim3(DEXT / 64, MT / 128, 1), blk, 0, stream>>>(
      xb, DIMc, 0, wkvab, DEXT, 1, 0, 0, scr, DEXT, 0, 0, DIMc);
  // 7) kext = [rmsnorm(c_kv) | rope(k_pe)]
  postkv_kernel<<<MT, blk, 0, stream>>>(scr, kv_norm_w, freqs, kextb);
  // 7b) kT = kext transposed per batch
  transpose_kext<<<dim3(Ss / 32, DEXT / 32, Bb), blk, 0, stream>>>(kextb, kTb);
  // 8) q_abs[h] = q_nope[h] @ W_nope[h]^T  (B elem = wkv_b[c*4096 + h*256 + d])
  gemm_bf16<<<dim3(KVLR / 64, MT / 128, Hh), blk, 0, stream>>>(
      qnopeb, Hh * DN, DN,
      wkvbb, 1, Hh * (DN + DV), 0, DN + DV,
      qextb, Hh * DEXT, DEXT, 1, DN);
  // 9) flash attention
  mla_flash<<<dim3(Ss / 16, Hh, Bb), blk, 0, stream>>>(qextb, kextb, kTb, obuf);
  // 10) o2[h] = o[h] @ W_v[h]^T  (B elem = wkv_b[c*4096 + h*256 + 128 + d])
  gemm_bf16<<<dim3(DV / 64, MT / 128, Hh), blk, 0, stream>>>(
      obuf, Hh * KVLR, KVLR,
      wkvbb, Hh * (DN + DV), 1, DN, DN + DV,
      ocat, Hh * DV, DV, 1, KVLR);
  // 11) out = ocat @ wo -> f32
  gemm_bf16<<<dim3(DIMc / 64, MT / 128, 1), blk, 0, stream>>>(
      ocat, Hh * DV, 0, wob, DIMc, 1, 0, 0, out, DIMc, 0, 0, Hh * DV);
}
